// DupchessNNUE_23132693856293
// MI455X (gfx1250) — matrix operand order repaired and checked by reading, not compiled, not run on hardware
//
#include <hip/hip_runtime.h>
#include <hip/hip_bf16.h>

// NNUE forward:  l1 = clip(X @ W_in^T + b_in), X = [wfeat|bfeat]  (the big op)
//                l2 = clip(l1 @ W1^T + b1); l3 = clip(l2 @ W2^T + b2); out = l3 @ W3^T + b3
//
// Roofline (MI455X): X stream = 1.34 GB fp32 from HBM = 57.6 us @ 23.3 TB/s.
// Layer-1 bf16 WMMA = 343.6 GFLOP. Secondary constraint: B (= W_in bf16, L2-resident)
// re-reads out of the 192 MB L2 -> wave tile 32x64 (8 WMMA per B-frag set) keeps that
// at 10.7 GB. 8 waves/WG cover N=512 so X leaves HBM exactly once.
// A is staged through a double-buffered LDS tile: 256 threads load each 32x64 fp32
// chunk once (instead of 8 waves redundantly), truncate to bf16 with v_perm_b32
// (exact: X is {0,1}), ds_store; waves build A fragments with ds_load_b128.
// 3-deep pipeline: global->regs (chunk i+2), regs->LDS (i+1), LDS+WMMA (i).
// Chunk barrier is a relaxed split-barrier (s_wait_dscnt 0 + s_barrier_signal/wait -1)
// so in-flight next-chunk B loads are NOT drained at the barrier (__syncthreads()
// would emit s_wait_loadcnt 0 and expose L2 latency 1280 times).

#define IN_FEATS 40960
#define K1 81920
#define L1 512
#define L2N 32
#define L3N 32
#define BATCH 4096

typedef __attribute__((ext_vector_type(16))) __bf16 v16bf;
typedef __attribute__((ext_vector_type(8)))  float  v8f;
typedef __attribute__((ext_vector_type(16))) unsigned short v16us;

union BFrag {
    v16bf bf;
    v16us us;
    unsigned short s[16];
};

__device__ __forceinline__ unsigned short f2bf(float f) {   // RNE, for weights/outputs
    unsigned int u = __float_as_uint(f);
    u += 0x7FFFu + ((u >> 16) & 1u);
    return (unsigned short)(u >> 16);
}

__device__ __forceinline__ float bf2f(unsigned short s) {
    return __uint_as_float(((unsigned int)s) << 16);
}

// pack hi16(x1):hi16(x0) in one v_perm_b32 (exact for {0,1} inputs)
__device__ __forceinline__ unsigned int pack_trunc(unsigned int x0, unsigned int x1) {
    return __builtin_amdgcn_perm(x1, x0, 0x07060302u);
}

// Workgroup barrier that orders LDS only: waits DScnt==0, then split-barrier.
// Deliberately does NOT drain loadcnt so in-flight global B loads cross the barrier.
__device__ __forceinline__ void lds_barrier() {
    asm volatile("s_wait_dscnt 0\n\t"
                 "s_barrier_signal -1\n\t"
                 "s_barrier_wait -1" ::: "memory");
}

// ---------------------------------------------------------------- fp32 -> bf16 (RNE)
__global__ void cvt_bf16_kernel(const float* __restrict__ in,
                                unsigned short* __restrict__ out, int n) {
    int i = (blockIdx.x * blockDim.x + threadIdx.x) * 8;
    if (i >= n) return;
    const float4* p = (const float4*)(in + i);
    float4 a = p[0], b = p[1];
    uint4 o;
    o.x = (unsigned)f2bf(a.x) | ((unsigned)f2bf(a.y) << 16);
    o.y = (unsigned)f2bf(a.z) | ((unsigned)f2bf(a.w) << 16);
    o.z = (unsigned)f2bf(b.x) | ((unsigned)f2bf(b.y) << 16);
    o.w = (unsigned)f2bf(b.z) | ((unsigned)f2bf(b.w) << 16);
    *(uint4*)(out + i) = o;
}

// ---------------------------------------------------------------- layer 1
__global__ void __launch_bounds__(256) nnue_layer1(
        const float* __restrict__ wfeat, const float* __restrict__ bfeat,
        const unsigned short* __restrict__ Wb, const float* __restrict__ b_in,
        unsigned short* __restrict__ l1b) {
    // LDS A tile: 2 buffers x 32 rows x 64 bf16, row stride 72 (144 B: 16B-aligned,
    // banks spread as 4*(9*row mod 16) -> conflict-free 16B fragment loads)
    __shared__ __align__(16) unsigned short lds[2][32][72];

    const int tid  = threadIdx.x;
    const int lane = tid & 31;
    const int wave = tid >> 5;
    const int m0   = blockIdx.x * 32;
    const int n0   = wave * 64;
    const int lm   = lane & 15;
    const int kh   = lane >> 4;

    // cooperative A loader coords: 8 threads per row, 8 K-elements per thread
    const int lrow = tid >> 3;          // 0..31
    const int lkof = (tid & 7) * 8;     // 0..56

    v8f acc[8] = {v8f{}, v8f{}, v8f{}, v8f{}, v8f{}, v8f{}, v8f{}, v8f{}};

    const unsigned short* wp0 = Wb + (size_t)(n0 + lm) * K1 + (size_t)kh * 16;
    const unsigned short* wp1 = wp0 + (size_t)16 * K1;
    const unsigned short* wp2 = wp0 + (size_t)32 * K1;
    const unsigned short* wp3 = wp0 + (size_t)48 * K1;

    v16us bqA[4], bqB[4];
    uint4 sq0, sq1;                     // cooperative A raw (chunk in flight)

    auto LDB = [&](v16us (&bq)[4], int k0) {
        bq[0] = *(const v16us*)(wp0 + k0);
        bq[1] = *(const v16us*)(wp1 + k0);
        bq[2] = *(const v16us*)(wp2 + k0);
        bq[3] = *(const v16us*)(wp3 + k0);
    };

    auto HALF = [&](int buf, int s, const v16us (&bq)[4]) {
        const int co = s * 32 + kh * 16;
        BFrag a0, a1;
        *(uint4*)&a0.s[0] = *(const uint4*)&lds[buf][lm][co];
        *(uint4*)&a0.s[8] = *(const uint4*)&lds[buf][lm][co + 8];
        *(uint4*)&a1.s[0] = *(const uint4*)&lds[buf][16 + lm][co];
        *(uint4*)&a1.s[8] = *(const uint4*)&lds[buf][16 + lm][co + 8];
        v16bf A0 = a0.bf, A1 = a1.bf;
        v16bf B0 = __builtin_bit_cast(v16bf, bq[0]);
        v16bf B1 = __builtin_bit_cast(v16bf, bq[1]);
        v16bf B2 = __builtin_bit_cast(v16bf, bq[2]);
        v16bf B3 = __builtin_bit_cast(v16bf, bq[3]);
        acc[0] = __builtin_amdgcn_wmma_f32_16x16x32_bf16(false, A0, false, B0, (short)0, acc[0], false, false);
        acc[1] = __builtin_amdgcn_wmma_f32_16x16x32_bf16(false, A0, false, B1, (short)0, acc[1], false, false);
        acc[2] = __builtin_amdgcn_wmma_f32_16x16x32_bf16(false, A0, false, B2, (short)0, acc[2], false, false);
        acc[3] = __builtin_amdgcn_wmma_f32_16x16x32_bf16(false, A0, false, B3, (short)0, acc[3], false, false);
        acc[4] = __builtin_amdgcn_wmma_f32_16x16x32_bf16(false, A1, false, B0, (short)0, acc[4], false, false);
        acc[5] = __builtin_amdgcn_wmma_f32_16x16x32_bf16(false, A1, false, B1, (short)0, acc[5], false, false);
        acc[6] = __builtin_amdgcn_wmma_f32_16x16x32_bf16(false, A1, false, B2, (short)0, acc[6], false, false);
        acc[7] = __builtin_amdgcn_wmma_f32_16x16x32_bf16(false, A1, false, B3, (short)0, acc[7], false, false);
    };

    // one contiguous feature half: xsrc = feat + (m0+lrow)*IN_FEATS + lkof, kb = W-K base
    auto RUN = [&](const float* xsrc, int kb) {
        auto SLOAD = [&](int ko) {
            sq0 = *(const uint4*)(xsrc + ko);
            sq1 = *(const uint4*)(xsrc + ko + 4);
            __builtin_prefetch(xsrc + ko + 1024, 0, 1);   // global_prefetch_b8 up the X stream
        };
        auto SSTORE = [&](int buf) {
            uint4 o;
            o.x = pack_trunc(sq0.x, sq0.y);
            o.y = pack_trunc(sq0.z, sq0.w);
            o.z = pack_trunc(sq1.x, sq1.y);
            o.w = pack_trunc(sq1.z, sq1.w);
            *(uint4*)&lds[buf][lrow][lkof] = o;
        };

        SLOAD(0);
        SSTORE(0);          // one prologue stall per half (2 total) - negligible
        SLOAD(64);
        LDB(bqA, kb);
        lds_barrier();

        #pragma unroll 1
        for (int ko = 0; ko < IN_FEATS; ko += 64) {
            const int cur = (ko >> 6) & 1;
            const int nxt = cur ^ 1;
            SSTORE(nxt);                                            // chunk ko+64 -> LDS
            SLOAD((ko + 128 < IN_FEATS) ? (ko + 128) : 0);          // chunk ko+128 -> regs
            LDB(bqB, kb + ko + 32);
            HALF(cur, 0, bqA);
            LDB(bqA, (ko + 64 < IN_FEATS) ? (kb + ko + 64) : kb);   // dummy reload on tail
            HALF(cur, 1, bqB);
            lds_barrier();
        }
    };

    RUN(wfeat + (size_t)(m0 + lrow) * IN_FEATS + lkof, 0);
    RUN(bfeat + (size_t)(m0 + lrow) * IN_FEATS + lkof, IN_FEATS);

    // C layout: VGPR r -> row base + r + 8*(lane>=16); col n0 + t*16 + (lane&15)
    #pragma unroll
    for (int f = 0; f < 2; ++f) {
        const int mrow = m0 + f * 16 + kh * 8;
        #pragma unroll
        for (int t = 0; t < 4; ++t) {
            const int n = n0 + t * 16 + lm;
            const float bias = b_in[n];
            #pragma unroll
            for (int r = 0; r < 8; ++r) {
                float v = acc[f * 4 + t][r] + bias;
                v = fminf(fmaxf(v, 0.0f), 1.0f);
                l1b[(size_t)(mrow + r) * L1 + n] = f2bf(v);
            }
        }
    }
}

// ---------------------------------------------------------------- layer 2 (K=512, N=32)
__global__ void __launch_bounds__(256) nnue_layer2(
        const unsigned short* __restrict__ l1b, const unsigned short* __restrict__ W1b,
        const float* __restrict__ b1, unsigned short* __restrict__ l2b) {
    const int lane = threadIdx.x & 31;
    const int wave = threadIdx.x >> 5;
    const int m0   = (blockIdx.x * 8 + wave) * 16;
    const int lm   = lane & 15;
    const int kh   = lane >> 4;

    v8f acc[2] = {v8f{}, v8f{}};
    const unsigned short* arow = l1b + (size_t)(m0 + lm) * L1;
    const unsigned short* w0   = W1b + (size_t)lm * L1 + kh * 16;
    const unsigned short* w1   = w0 + 16 * L1;

    #pragma unroll
    for (int k0 = 0; k0 < L1; k0 += 32) {
        BFrag a;
        *(uint4*)&a.s[0] = *(const uint4*)(arow + k0 + kh * 16);
        *(uint4*)&a.s[8] = *(const uint4*)(arow + k0 + 8 + kh * 16);
        BFrag b0, b1f;
        b0.us  = *(const v16us*)(w0 + k0);
        b1f.us = *(const v16us*)(w1 + k0);
        acc[0] = __builtin_amdgcn_wmma_f32_16x16x32_bf16(false, a.bf, false, b0.bf,  (short)0, acc[0], false, false);
        acc[1] = __builtin_amdgcn_wmma_f32_16x16x32_bf16(false, a.bf, false, b1f.bf, (short)0, acc[1], false, false);
    }

    const int mrow = m0 + kh * 8;
    #pragma unroll
    for (int t = 0; t < 2; ++t) {
        const int n = t * 16 + lm;
        const float bias = b1[n];
        #pragma unroll
        for (int r = 0; r < 8; ++r) {
            float v = acc[t][r] + bias;
            v = fminf(fmaxf(v, 0.0f), 1.0f);
            l2b[(size_t)(mrow + r) * L2N + n] = f2bf(v);
        }
    }
}

// ---------------------------------------------------------------- layer 3 (K=32, N=32)
__global__ void __launch_bounds__(256) nnue_layer3(
        const unsigned short* __restrict__ l2b, const unsigned short* __restrict__ W2b,
        const float* __restrict__ b2, unsigned short* __restrict__ l3b) {
    const int lane = threadIdx.x & 31;
    const int wave = threadIdx.x >> 5;
    const int m0   = (blockIdx.x * 8 + wave) * 16;
    const int lm   = lane & 15;
    const int kh   = lane >> 4;

    const unsigned short* arow = l2b + (size_t)(m0 + lm) * L2N;
    BFrag a;
    *(uint4*)&a.s[0] = *(const uint4*)(arow + kh * 16);
    *(uint4*)&a.s[8] = *(const uint4*)(arow + 8 + kh * 16);

    const unsigned short* w0 = W2b + (size_t)lm * L3N + kh * 16;
    BFrag b0, b1f;
    b0.us  = *(const v16us*)(w0);
    b1f.us = *(const v16us*)(w0 + 16 * L3N);

    v8f acc0 = {}, acc1 = {};
    acc0 = __builtin_amdgcn_wmma_f32_16x16x32_bf16(false, a.bf, false, b0.bf,  (short)0, acc0, false, false);
    acc1 = __builtin_amdgcn_wmma_f32_16x16x32_bf16(false, a.bf, false, b1f.bf, (short)0, acc1, false, false);

    const int mrow = m0 + kh * 8;
    #pragma unroll
    for (int r = 0; r < 8; ++r) {
        float v0 = acc0[r] + b2[lm];
        float v1 = acc1[r] + b2[16 + lm];
        v0 = fminf(fmaxf(v0, 0.0f), 1.0f);
        v1 = fminf(fmaxf(v1, 0.0f), 1.0f);
        l3b[(size_t)(mrow + r) * L3N + lm]      = f2bf(v0);
        l3b[(size_t)(mrow + r) * L3N + 16 + lm] = f2bf(v1);
    }
}

// ---------------------------------------------------------------- output layer
__global__ void nnue_out(const unsigned short* __restrict__ l3b,
                         const float* __restrict__ W3, const float* __restrict__ b3,
                         float* __restrict__ out) {
    int m = blockIdx.x * blockDim.x + threadIdx.x;
    if (m >= BATCH) return;
    const unsigned short* r = l3b + (size_t)m * L3N;
    float s = 0.0f;
    #pragma unroll
    for (int j = 0; j < L3N; ++j) s += bf2f(r[j]) * W3[j];
    out[m] = s + b3[0];
}

// ---------------------------------------------------------------- launcher
extern "C" void kernel_launch(void* const* d_in, const int* in_sizes, int n_in,
                              void* d_out, int out_size, void* d_ws, size_t ws_size,
                              hipStream_t stream) {
    const float* wfeat = (const float*)d_in[0];
    const float* bfeat = (const float*)d_in[1];
    const float* W_in  = (const float*)d_in[2];
    const float* b_in  = (const float*)d_in[3];
    const float* W1    = (const float*)d_in[4];
    const float* b1    = (const float*)d_in[5];
    const float* W2    = (const float*)d_in[6];
    const float* b2    = (const float*)d_in[7];
    const float* W3    = (const float*)d_in[8];
    const float* b3    = (const float*)d_in[9];
    float* out = (float*)d_out;

    char* ws = (char*)d_ws;
    unsigned short* Wb  = (unsigned short*)(ws);                 // 512*81920*2 = 83,886,080
    unsigned short* W1b = (unsigned short*)(ws + 83886080);      // 32*512*2    =     32,768
    unsigned short* W2b = (unsigned short*)(ws + 83918848);      // 32*32*2     =      2,048
    unsigned short* l1b = (unsigned short*)(ws + 83920896);      // 4096*512*2  =  4,194,304
    unsigned short* l2b = (unsigned short*)(ws + 88115200);      // 4096*32*2   =    262,144
    unsigned short* l3b = (unsigned short*)(ws + 88377344);      // 4096*32*2   =    262,144

    // fp32 -> bf16 weight staging (W_in becomes 84 MB: L2-resident on 192 MB L2)
    cvt_bf16_kernel<<<20480, 256, 0, stream>>>(W_in, Wb, 512 * K1);
    cvt_bf16_kernel<<<8,     256, 0, stream>>>(W1,   W1b, 32 * 512);
    cvt_bf16_kernel<<<1,     256, 0, stream>>>(W2,   W2b, 32 * 32);

    nnue_layer1<<<BATCH / 32,  256, 0, stream>>>(wfeat, bfeat, Wb, b_in, l1b);
    nnue_layer2<<<BATCH / 128, 256, 0, stream>>>(l1b, W1b, b1, l2b);
    nnue_layer3<<<BATCH / 128, 256, 0, stream>>>(l2b, W2b, b2, l3b);
    nnue_out<<<BATCH / 256,    256, 0, stream>>>(l3b, W3, b3, out);
}
